// BaseLSGSelfAttention_18734647345539
// MI455X (gfx1250) — matrix operand order repaired and checked by reading, not compiled, not run on hardware
//
#include <hip/hip_runtime.h>

typedef __attribute__((ext_vector_type(16))) _Float16 v16h;
typedef __attribute__((ext_vector_type(8)))  _Float16 v8h;
typedef __attribute__((ext_vector_type(2)))  _Float16 v2h;
typedef __attribute__((ext_vector_type(8)))  float    v8f;

#define TT 4096
#define DD 64
#define BLK 128
#define NB 32
#define NEG_BIG (-1.0e30f)
#define LOG2E 1.4426950408889634f

static __device__ __forceinline__ float xor16(float x) {
  return __shfl_xor(x, 16, 32);
}

static __device__ __forceinline__ int clampT(int s) {
  return s < 0 ? 0 : (s > TT - 1 ? TT - 1 : s);
}

// One workgroup = (n*h, qblock). 256 threads = 8 waves; each wave owns 16 query rows.
// Flash-attention with Sᵀ = K·Qᵀ and Oᵀ = Vᵀ·P so softmax state is per-lane (query = lane%16).
// Double-buffered LDS staging, software-pipelined so global loads overlap the WMMAs.
__global__ __launch_bounds__(256)
void lsg_attn_main(const float* __restrict__ Q, const float* __restrict__ K,
                   const float* __restrict__ V, const float* __restrict__ AM,
                   float* __restrict__ O) {
  __shared__ _Float16 ksh[2][32 * DD];   // K tile, row-major [key][d], f16
  __shared__ _Float16 vsh[2][DD * 32];   // V tile, transposed [d][key], f16

  const int bid  = blockIdx.x;
  const int qb   = bid & (NB - 1);
  const int nh   = bid >> 5;
  const int nidx = nh >> 4;
  const size_t base = (size_t)nh * TT * DD;
  const float* Qb = Q + base;
  const float* Kb = K + base;
  const float* Vb = V + base;
  const float* AMb = AM + (size_t)nidx * TT;
  float* Ob = O + base;

  const int tid  = threadIdx.x;
  const int lane = tid & 31;
  const int wv   = tid >> 5;
  const int col  = lane & 15;   // query column of this lane
  const int g    = lane >> 4;   // half-wave group

  const float scale2 = 0.125f * LOG2E;   // (1/sqrt(64)) * log2(e): softmax in base 2

  // ---- Preload Q as B-tiles of Qᵀ (32d x 16q). B layout: lane holds d = 32c + 16g + i.
  // Also accumulate the global-key dot q·k0 on the fly (same element pattern).
  const int qrow = qb * BLK + wv * 16 + col;
  v16h qB[2];
  float gdot = 0.0f;
  #pragma unroll
  for (int c = 0; c < 2; ++c) {
    const int d0 = 32 * c + 16 * g;
    const float* qp = Qb + (size_t)qrow * DD + d0;
    const float* kp = Kb + d0;                 // K row 0 (global key)
    v16h t;
    #pragma unroll
    for (int i = 0; i < 16; ++i) {
      float qv = qp[i];
      gdot += qv * kp[i];
      t[i] = (_Float16)qv;
    }
    qB[c] = t;
  }
  gdot += xor16(gdot);

  // Online-softmax state (base-2 domain), seeded with the global key (p_g == 1).
  float m    = gdot * scale2;                  // global-key mask is 0 in reference
  float lsum = 1.0f;

  // Oᵀ accumulators: 4 tiles of (16d x 16q); D layout row = d-in-tile = v + 8g.
  // Init = 1.0 * v0 broadcast across queries.
  v8f o[4];
  #pragma unroll
  for (int dt = 0; dt < 4; ++dt) {
    const float* vp = Vb + 16 * dt + 8 * g;
    #pragma unroll
    for (int v = 0; v < 8; ++v) o[dt][v] = vp[v];
  }

  const int s0 = qb * BLK - BLK;               // first local key position

  // ---- Staging thread roles (fixed across iterations)
  const int kk  = tid >> 3;                    // key row for K staging
  const int kd0 = (tid & 7) * 8;               // d offset for K staging
  const int kpair = tid & 15;                  // key pair for V staging
  const int dq    = tid >> 4;                  // d quad for V staging

  float kst[8];                                // staged K row segment (f32)
  float vstA[4], vstB[4];                      // staged V elements for 2 keys

  auto loadStage = [&](int it) {
    const int kw = 32 * it;
    const float* kp = Kb + (size_t)clampT(s0 + kw + kk) * DD + kd0;
    #pragma unroll
    for (int i = 0; i < 8; ++i) kst[i] = kp[i];
    const float* va = Vb + (size_t)clampT(s0 + kw + 2 * kpair) * DD + 4 * dq;
    const float* vb = Vb + (size_t)clampT(s0 + kw + 2 * kpair + 1) * DD + 4 * dq;
    #pragma unroll
    for (int j = 0; j < 4; ++j) { vstA[j] = va[j]; vstB[j] = vb[j]; }
  };

  auto storeStage = [&](int buf) {
    v8h kh;
    #pragma unroll
    for (int i = 0; i < 8; ++i) kh[i] = (_Float16)kst[i];
    *(v8h*)&ksh[buf][kk * DD + kd0] = kh;
    #pragma unroll
    for (int j = 0; j < 4; ++j) {
      v2h pk; pk[0] = (_Float16)vstA[j]; pk[1] = (_Float16)vstB[j];
      *(v2h*)&vsh[buf][(4 * dq + j) * 32 + 2 * kpair] = pk;
    }
  };

  // Prologue: fill buffer 0 for iteration 0.
  loadStage(0);
  storeStage(0);

  for (int it = 0; it < 12; ++it) {            // 12 * 32 = 384 local keys
    __syncthreads();                           // buffer it&1 visible to all waves
    const int buf = it & 1;
    const bool more = it < 11;
    if (more) loadStage(it + 1);               // global loads overlap the WMMAs below

    const int kw = 32 * it;

    // ---- Sᵀ tiles: two 16-key tiles, each accumulated over d in 2 WMMAs (k=32).
    float p0[8], p1[8];
    float lmax = NEG_BIG;
    #pragma unroll
    for (int kt = 0; kt < 2; ++kt) {
      v8f s8 = {};
      #pragma unroll
      for (int c = 0; c < 2; ++c) {
        const int krow = 16 * kt + col;        // A row = key-in-tile
        const int doff = 32 * c + 16 * g;      // A layout: d = doff+{0..15}
        v8h lo = *(const v8h*)&ksh[buf][krow * DD + doff];
        v8h hi = *(const v8h*)&ksh[buf][krow * DD + doff + 8];
        v16h a = __builtin_shufflevector(lo, hi, 0,1,2,3,4,5,6,7,8,9,10,11,12,13,14,15);
        s8 = __builtin_amdgcn_wmma_f32_16x16x32_f16(false, a, false, qB[c],
                                                    (short)0, s8, false, false);
      }
      float* pp = kt ? p1 : p0;
      #pragma unroll
      for (int v = 0; v < 8; ++v) {
        const int skey = s0 + kw + 16 * kt + 8 * g + v;  // D row = v + 8g
        float sv = NEG_BIG;
        if (skey >= 1 && skey < TT) sv = s8[v] * scale2 + AMb[skey] * LOG2E;
        pp[v] = sv;
        lmax = fmaxf(lmax, sv);
      }
    }

    // ---- Online softmax update (state per lane, query = col), base-2 domain.
    float mnew  = fmaxf(m, fmaxf(lmax, xor16(lmax)));
    float alpha = __builtin_exp2f(m - mnew);
    float ls = 0.0f;
    #pragma unroll
    for (int v = 0; v < 8; ++v) {
      p0[v] = __builtin_exp2f(p0[v] - mnew);
      p1[v] = __builtin_exp2f(p1[v] - mnew);
      ls += p0[v] + p1[v];
    }
    lsum = lsum * alpha + ls + xor16(ls);
    m = mnew;
    #pragma unroll
    for (int dt = 0; dt < 4; ++dt)
      #pragma unroll
      for (int v = 0; v < 8; ++v) o[dt][v] *= alpha;

    // ---- Pack P into B layout (lane needs keys 16g+0..15): cross-half exchange.
    v16h pB;
    #pragma unroll
    for (int v = 0; v < 8; ++v) {
      float x0 = xor16(p0[v]);
      float x1 = xor16(p1[v]);
      pB[v]     = (_Float16)(g == 0 ? p0[v] : x1);
      pB[v + 8] = (_Float16)(g == 0 ? x0 : p1[v]);
    }

    // ---- Oᵀ += Vᵀ · P  (4 d-tiles).
    #pragma unroll
    for (int dt = 0; dt < 4; ++dt) {
      const int drow = 16 * dt + col;          // A row = d
      v8h lo = *(const v8h*)&vsh[buf][drow * 32 + 16 * g];
      v8h hi = *(const v8h*)&vsh[buf][drow * 32 + 16 * g + 8];
      v16h a = __builtin_shufflevector(lo, hi, 0,1,2,3,4,5,6,7,8,9,10,11,12,13,14,15);
      o[dt] = __builtin_amdgcn_wmma_f32_16x16x32_f16(false, a, false, pB,
                                                     (short)0, o[dt], false, false);
    }

    // ---- Convert + store next tile into the other buffer (load-wait lands here,
    // after the WMMAs; visibility guaranteed by the barrier at top of next iter).
    if (more) storeStage(buf ^ 1);
  }

  // ---- Normalize and store: lane writes 8 contiguous d per tile for query `qrow`.
  const float inv = 1.0f / lsum;
  #pragma unroll
  for (int dt = 0; dt < 4; ++dt) {
    float4* op = (float4*)(Ob + (size_t)qrow * DD + 16 * dt + 8 * g);
    float4 r0, r1;
    r0.x = o[dt][0] * inv; r0.y = o[dt][1] * inv;
    r0.z = o[dt][2] * inv; r0.w = o[dt][3] * inv;
    r1.x = o[dt][4] * inv; r1.y = o[dt][5] * inv;
    r1.z = o[dt][6] * inv; r1.w = o[dt][7] * inv;
    op[0] = r0; op[1] = r1;
  }
}

// BOS row: query 0 attends to all 4096 keys with the raw mask. One WG per (n*h).
__global__ __launch_bounds__(256)
void lsg_attn_bos(const float* __restrict__ Q, const float* __restrict__ K,
                  const float* __restrict__ V, const float* __restrict__ AM,
                  float* __restrict__ O) {
  __shared__ float sq[DD];
  __shared__ float sp[TT];
  __shared__ float red[8];
  __shared__ float pacc[4][DD];

  const int nh   = blockIdx.x;
  const int nidx = nh >> 4;
  const size_t base = (size_t)nh * TT * DD;
  const float* Qb = Q + base;
  const float* Kb = K + base;
  const float* Vb = V + base;
  const float* AMb = AM + (size_t)nidx * TT;

  const int tid = threadIdx.x, lane = tid & 31, wv = tid >> 5;

  if (tid < DD) sq[tid] = Qb[tid];
  __syncthreads();

  float sc[16];
  float lmax = NEG_BIG;
  for (int i = 0; i < 16; ++i) {
    const int j = tid + 256 * i;
    const float* kp = Kb + (size_t)j * DD;
    float acc = 0.0f;
    for (int d = 0; d < DD; ++d) acc += sq[d] * kp[d];
    acc = acc * 0.125f + AMb[j];
    sc[i] = acc;
    lmax = fmaxf(lmax, acc);
  }
  for (int off = 16; off >= 1; off >>= 1) lmax = fmaxf(lmax, __shfl_xor(lmax, off, 32));
  if (lane == 0) red[wv] = lmax;
  __syncthreads();
  float mmax = red[0];
  for (int w = 1; w < 8; ++w) mmax = fmaxf(mmax, red[w]);

  float lsum = 0.0f;
  for (int i = 0; i < 16; ++i) {
    const int j = tid + 256 * i;
    const float p = __expf(sc[i] - mmax);
    sp[j] = p;
    lsum += p;
  }
  for (int off = 16; off >= 1; off >>= 1) lsum += __shfl_xor(lsum, off, 32);
  __syncthreads();
  if (lane == 0) red[wv] = lsum;
  __syncthreads();
  float ell = 0.0f;
  for (int w = 0; w < 8; ++w) ell += red[w];

  const int d = tid & 63, grp = tid >> 6;
  float acc = 0.0f;
  for (int j = grp; j < TT; j += 4) acc += sp[j] * Vb[(size_t)j * DD + d];
  pacc[grp][d] = acc;
  __syncthreads();
  if (tid < DD) {
    O[base + tid] = (pacc[0][tid] + pacc[1][tid] + pacc[2][tid] + pacc[3][tid]) / ell;
  }
}

extern "C" void kernel_launch(void* const* d_in, const int* in_sizes, int n_in,
                              void* d_out, int out_size, void* d_ws, size_t ws_size,
                              hipStream_t stream) {
  const float* Q  = (const float*)d_in[0];
  const float* K  = (const float*)d_in[1];
  const float* V  = (const float*)d_in[2];
  const float* AM = (const float*)d_in[3];
  float* O = (float*)d_out;

  // 32 (n*h) pairs x 32 query blocks of 128
  lsg_attn_main<<<dim3(32 * 32), dim3(256), 0, stream>>>(Q, K, V, AM, O);
  // BOS row overwrite must follow (stream-ordered, matching reference .at[:, :, :1].set)
  lsg_attn_bos<<<dim3(32), dim3(256), 0, stream>>>(Q, K, V, AM, O);
}
